// SDCN_22204980920337
// MI455X (gfx1250) — matrix operand order repaired and checked by reading, not compiled, not run on hardware
//
#include <hip/hip_runtime.h>
#include <math.h>

// ---------------------------------------------------------------------------
// Types for CDNA5 WMMA (wave32): D(f32 16x16) = A(bf16 16x32) x B(bf16 32x16) + C
// ---------------------------------------------------------------------------
typedef __attribute__((ext_vector_type(16))) __bf16 v16bf;
typedef __attribute__((ext_vector_type(8)))  float  v8f;

union FragBF { unsigned int u[8]; v16bf v; };
union AccU   { v8f v; float f[8]; };
union V4U    { uint4 v; unsigned short s[8]; };

__device__ __forceinline__ unsigned short f2bf(float f) {
  unsigned int u = __builtin_bit_cast(unsigned int, f);
  unsigned int r = u + 0x7FFFu + ((u >> 16) & 1u);   // round-to-nearest-even
  return (unsigned short)(r >> 16);
}

// ---------------------------------------------------------------------------
// Tiled bf16 WMMA GEMM: C[M,N](f32,ldc) = A[.,Kloop](bf16,lda) @ B[Kloop,.](bf16,ldb)
// A rows padded to gridDim.y*128, B rows >= Kloop, B cols >= gridDim.x*64,
// lda/ldb multiples of 8, Kloop/chunk multiples of 64.
// flags: bit0 = ReLU epilogue, bit1 = split-K atomic accumulate (no bias/relu).
// blockIdx.z selects K-chunk [z*chunk, min((z+1)*chunk, Kloop)).
// ---------------------------------------------------------------------------
#define BM 128
#define BN 64
#define BK 64
#define LDSA 72   // ushort stride (144B rows: 16B-aligned b128, staggered banks)
#define LDSB 72

__global__ __launch_bounds__(256) void sdcn_gemm_bf16(
    const unsigned short* __restrict__ A, int lda,
    const unsigned short* __restrict__ B, int ldb,
    const float* __restrict__ bias,
    float* __restrict__ C, int ldc,
    int M, int N, int Kloop, int flags, int chunk) {
  __shared__ unsigned short As[BM * LDSA];
  __shared__ unsigned short Bs[BN * LDSB];

  const int tid  = threadIdx.x;
  const int lane = tid & 31;
  const int wid  = tid >> 5;
  const int wm   = wid & 3;        // wave row (0..3)
  const int wn   = wid >> 2;       // wave col (0..1)
  const int bm   = blockIdx.y * BM;
  const int bn   = blockIdx.x * BN;

  const int arow  = tid >> 1;            // A tile: 128 rows, 2 threads/row
  const int ahalf = (tid & 1) * 32;      // 32 bf16 per thread (4x B128)
  const int bkrow = tid >> 2;            // B tile: 64 k-rows, 4 threads/row
  const int bnch  = (tid & 3) * 16;      // 16 bf16 per thread (2x B128)

  const int mhalf = lane >> 4;           // lane half (ISA wave32 layouts)
  const int ml    = lane & 15;

  v8f acc[2][2];
  #pragma unroll
  for (int a = 0; a < 2; ++a)
    #pragma unroll
    for (int b = 0; b < 2; ++b)
      #pragma unroll
      for (int r = 0; r < 8; ++r) acc[a][b][r] = 0.0f;

  const int kb = blockIdx.z * chunk;
  int ke = kb + chunk; if (ke > Kloop) ke = Kloop;

  for (int k0 = kb; k0 < ke; k0 += BK) {
    // ---- global loads (B128) ----
    const uint4* ap = (const uint4*)(A + (size_t)(bm + arow) * lda + k0 + ahalf);
    uint4 av0 = ap[0], av1 = ap[1], av2 = ap[2], av3 = ap[3];
    V4U bv0, bv1;
    const uint4* bp = (const uint4*)(B + (size_t)(k0 + bkrow) * ldb + bn + bnch);
    bv0.v = bp[0]; bv1.v = bp[1];
    if (k0 + BK < ke) {
      __builtin_prefetch(A + (size_t)(bm + arow) * lda + k0 + BK + ahalf, 0, 1);
      __builtin_prefetch(B + (size_t)(k0 + BK + bkrow) * ldb + bn + bnch, 0, 1);
    }

    __syncthreads();                       // previous tile fully consumed
    uint4* asp = (uint4*)&As[arow * LDSA + ahalf];
    asp[0] = av0; asp[1] = av1; asp[2] = av2; asp[3] = av3;
    #pragma unroll
    for (int e = 0; e < 8; ++e) {          // store B transposed: Bs[n][k]
      Bs[(bnch + e)     * LDSB + bkrow] = bv0.s[e];
      Bs[(bnch + 8 + e) * LDSB + bkrow] = bv1.s[e];
    }
    __syncthreads();

    // ---- two 16x16x32 K-steps per staged tile ----
    #pragma unroll
    for (int s = 0; s < 2; ++s) {
      FragBF afr[2], bfr[2];
      #pragma unroll
      for (int f = 0; f < 2; ++f) {
        const int rbase = (wm * 32 + f * 16 + ml) * LDSA + s * 32;
        #pragma unroll
        for (int i = 0; i < 8; ++i) {
          // A 16x32 bf16: VGPR i holds K = (i/4)*16 + half*8 + (i%4)*2 (+1)
          const int kk = ((i >> 2) << 4) + (mhalf << 3) + ((i & 3) << 1);
          afr[f].u[i] = *(const unsigned int*)&As[rbase + kk];
        }
        const int nbase = (wn * 32 + f * 16 + ml) * LDSB + s * 32 + (mhalf << 3);
        #pragma unroll
        for (int i = 0; i < 8; ++i)
          // B 32x16 bf16: lane=N, VGPR i holds K = (i/4)*16 + half*8 + (i%4)*2 (+1)
          bfr[f].u[i] = *(const unsigned int*)&Bs[nbase + ((i >> 2) << 4) + ((i & 3) << 1)];
      }
      // zig-zag issue with ISA reuse hints (current op identical to previous)
      acc[0][0] = __builtin_amdgcn_wmma_f32_16x16x32_bf16(
          false, afr[0].v, false, bfr[0].v, (short)0, acc[0][0], false, false);
      acc[0][1] = __builtin_amdgcn_wmma_f32_16x16x32_bf16(
          false, afr[0].v, false, bfr[1].v, (short)0, acc[0][1], true,  false);
      acc[1][1] = __builtin_amdgcn_wmma_f32_16x16x32_bf16(
          false, afr[1].v, false, bfr[1].v, (short)0, acc[1][1], false, true);
      acc[1][0] = __builtin_amdgcn_wmma_f32_16x16x32_bf16(
          false, afr[1].v, false, bfr[0].v, (short)0, acc[1][0], true,  false);
    }
  }

  // ---- epilogue: C/D f32 layout: VGPR r -> M = r + half*8, lane%16 = N ----
  #pragma unroll
  for (int fm = 0; fm < 2; ++fm) {
    #pragma unroll
    for (int fn = 0; fn < 2; ++fn) {
      const int col = bn + wn * 32 + fn * 16 + ml;
      if (col >= N) continue;
      const float bvv = bias ? bias[col] : 0.0f;
      AccU au; au.v = acc[fm][fn];
      #pragma unroll
      for (int r = 0; r < 8; ++r) {
        const int row = bm + wm * 32 + fm * 16 + mhalf * 8 + r;
        if (row < M) {
          if (flags & 2) {
            atomicAdd(&C[(size_t)row * ldc + col], au.f[r]);
          } else {
            float v = au.f[r] + bvv;
            if (flags & 1) v = v > 0.0f ? v : 0.0f;
            C[(size_t)row * ldc + col] = v;
          }
        }
      }
    }
  }
}

// ---------------------------------------------------------------------------
// helpers: zero fill, bias/activation epilogue (for split-K outputs)
// ---------------------------------------------------------------------------
__global__ void sdcn_zerof(float* __restrict__ p, size_t n) {
  size_t i = (size_t)blockIdx.x * blockDim.x + threadIdx.x;
  if (i < n) p[i] = 0.0f;
}

__global__ void sdcn_bias_act(float* __restrict__ C, int M, int N,
                              const float* __restrict__ bias, int relu) {
  size_t i = (size_t)blockIdx.x * blockDim.x + threadIdx.x;
  if (i >= (size_t)M * N) return;
  int j = (int)(i % (size_t)N);
  float v = C[i] + (bias ? bias[j] : 0.0f);
  if (relu) v = v > 0.0f ? v : 0.0f;
  C[i] = v;
}

// ---------------------------------------------------------------------------
// f32 -> zero-padded bf16
// ---------------------------------------------------------------------------
__global__ void sdcn_cvt_pad(const float* __restrict__ src, int M, int N,
                             unsigned short* __restrict__ dst, int Mp, int Np) {
  size_t idx = (size_t)blockIdx.x * blockDim.x + threadIdx.x;
  size_t total = (size_t)Mp * (size_t)Np;
  if (idx >= total) return;
  int j = (int)(idx % (size_t)Np);
  int i = (int)(idx / (size_t)Np);
  float v = (i < M && j < N) ? src[(size_t)i * N + j] : 0.0f;
  dst[idx] = f2bf(v);
}

// ---------------------------------------------------------------------------
// gate: P = l2norm_rows(softmax(leaky_relu(concat(X0..X4) @ W + b))) ; P ld=8
// one wave per row
// ---------------------------------------------------------------------------
__global__ __launch_bounds__(256) void sdcn_gate_l2(
    const float* __restrict__ X0, int d0, const float* __restrict__ X1, int d1,
    const float* __restrict__ X2, int d2, const float* __restrict__ X3, int d3,
    const float* __restrict__ X4, int d4,
    const float* __restrict__ W, const float* __restrict__ bias,
    int G, int Nrows, float* __restrict__ P) {
  const int lane = threadIdx.x & 31;
  const int row  = blockIdx.x * (blockDim.x >> 5) + (threadIdx.x >> 5);
  if (row >= Nrows) return;
  float g[8];
  #pragma unroll
  for (int j = 0; j < 8; ++j) g[j] = 0.0f;
  const float* Xs[5] = {X0, X1, X2, X3, X4};
  const int    ds[5] = {d0, d1, d2, d3, d4};
  int off = 0;
  for (int p = 0; p < 5; ++p) {
    const float* X = Xs[p]; const int dp = ds[p];
    if (X == nullptr || dp == 0) continue;
    for (int c = lane; c < dp; c += 32) {
      const float x = X[(size_t)row * dp + c];
      const float* wr = W + (size_t)(off + c) * G;
      for (int j = 0; j < G; ++j) g[j] += x * wr[j];
    }
    off += dp;
  }
  for (int j = 0; j < G; ++j)
    for (int s = 16; s > 0; s >>= 1) g[j] += __shfl_xor(g[j], s, 32);
  float m = -3.4e38f;
  for (int j = 0; j < G; ++j) {
    float v = g[j] + bias[j];
    v = v > 0.0f ? v : 0.01f * v;               // leaky_relu slope 0.01
    g[j] = v; m = fmaxf(m, v);
  }
  float s = 0.0f;
  for (int j = 0; j < G; ++j) { g[j] = expf(g[j] - m); s += g[j]; }
  float n2 = 0.0f;
  for (int j = 0; j < G; ++j) { g[j] /= s; n2 += g[j] * g[j]; }
  float nrm = sqrtf(n2); nrm = nrm > 1e-12f ? nrm : 1e-12f;
  if (lane == 0)
    for (int j = 0; j < G; ++j) P[(size_t)row * 8 + j] = g[j] / nrm;
}

// mix -> padded bf16: dst = bf16(P[:,0]*Z + P[:,1]*H)
__global__ void sdcn_mix_bf16(const float* __restrict__ P, const float* __restrict__ Z,
                              const float* __restrict__ H, int M, int N,
                              unsigned short* __restrict__ dst, int Mp, int Np) {
  size_t idx = (size_t)blockIdx.x * blockDim.x + threadIdx.x;
  size_t total = (size_t)Mp * (size_t)Np;
  if (idx >= total) return;
  int j = (int)(idx % (size_t)Np);
  int i = (int)(idx / (size_t)Np);
  float v = 0.0f;
  if (i < M && j < N)
    v = P[(size_t)i * 8 + 0] * Z[(size_t)i * N + j] +
        P[(size_t)i * 8 + 1] * H[(size_t)i * N + j];
  dst[idx] = f2bf(v);
}

// net = [w0*z1 | w1*z2 | w2*z3 | w3*z4 | w4*z] -> padded bf16 (cols: 500/500/2000/10/10)
__global__ void sdcn_build_net(const float* __restrict__ W5,
                               const float* __restrict__ z1, const float* __restrict__ z2,
                               const float* __restrict__ z3, const float* __restrict__ z4,
                               const float* __restrict__ z, int M,
                               unsigned short* __restrict__ dst, int Mp, int Np) {
  size_t idx = (size_t)blockIdx.x * blockDim.x + threadIdx.x;
  size_t total = (size_t)Mp * (size_t)Np;
  if (idx >= total) return;
  int j = (int)(idx % (size_t)Np);
  int i = (int)(idx / (size_t)Np);
  float v = 0.0f;
  if (i < M && j < 3020) {
    const float* w = W5 + (size_t)i * 8;
    if      (j < 500)  v = w[0] * z1[(size_t)i * 500  + j];
    else if (j < 1000) v = w[1] * z2[(size_t)i * 500  + (j - 500)];
    else if (j < 3000) v = w[2] * z3[(size_t)i * 2000 + (j - 1000)];
    else if (j < 3010) v = w[3] * z4[(size_t)i * 10   + (j - 3000)];
    else               v = w[4] * z [(size_t)i * 10   + (j - 3010)];
  }
  dst[idx] = f2bf(v);
}

__global__ void sdcn_softmax_rows(const float* __restrict__ in, float* __restrict__ out,
                                  int M, int C) {
  int i = blockIdx.x * blockDim.x + threadIdx.x;
  if (i >= M) return;
  float m = -3.4e38f;
  for (int j = 0; j < C; ++j) m = fmaxf(m, in[(size_t)i * C + j]);
  float s = 0.0f;
  for (int j = 0; j < C; ++j) s += expf(in[(size_t)i * C + j] - m);
  for (int j = 0; j < C; ++j) out[(size_t)i * C + j] = expf(in[(size_t)i * C + j] - m) / s;
}

// student-t soft assignment (v=1): q_ij = (1+||z_i-mu_j||^2)^-1, row-normalized
__global__ void sdcn_student_t(const float* __restrict__ z, const float* __restrict__ cl,
                               float* __restrict__ q, int M) {
  int i = blockIdx.x * blockDim.x + threadIdx.x;
  if (i >= M) return;
  float zr[10];
  #pragma unroll
  for (int d = 0; d < 10; ++d) zr[d] = z[(size_t)i * 10 + d];
  float s = 0.0f, qq[20];
  for (int j = 0; j < 20; ++j) {
    float d2 = 0.0f;
    #pragma unroll
    for (int d = 0; d < 10; ++d) { float df = zr[d] - cl[j * 10 + d]; d2 += df * df; }
    qq[j] = 1.0f / (1.0f + d2); s += qq[j];
  }
  for (int j = 0; j < 20; ++j) q[(size_t)i * 20 + j] = qq[j] / s;
}

__global__ void sdcn_zero1(float* p) {
  if (threadIdx.x == 0 && blockIdx.x == 0) *p = 0.0f;
}

// z_F = softmax(pzh0*predict + pzh1*q) ; pseudo-label loss reduction
__global__ void sdcn_fuse_loss(const float* __restrict__ P, const float* __restrict__ pred,
                               const float* __restrict__ q, float* __restrict__ zF,
                               float* __restrict__ loss, int M) {
  int i = blockIdx.x * blockDim.x + threadIdx.x;
  if (i >= M) return;
  const float p0 = P[(size_t)i * 8 + 0], p1 = P[(size_t)i * 8 + 1];
  float t[20], m = -3.4e38f;
  for (int j = 0; j < 20; ++j) {
    t[j] = p0 * pred[(size_t)i * 20 + j] + p1 * q[(size_t)i * 20 + j];
    m = fmaxf(m, t[j]);
  }
  float s = 0.0f;
  for (int j = 0; j < 20; ++j) { t[j] = expf(t[j] - m); s += t[j]; }
  float n2 = 0.0f; int amax = 0; float best = -1.0f;
  for (int j = 0; j < 20; ++j) {
    t[j] /= s;
    zF[(size_t)i * 20 + j] = t[j];
    n2 += t[j] * t[j];
    if (t[j] > best) { best = t[j]; amax = j; }   // first max (jnp.argmax)
  }
  float nrm = sqrtf(n2); nrm = nrm > 1e-12f ? nrm : 1e-12f;
  float acc = 0.0f;
  for (int j = 0; j < 20; ++j) {
    const float wl = (t[j] / nrm >= 0.7f) ? 1.0f : 0.0f;
    const float oh = (j == amax) ? 1.0f : 0.0f;
    acc += wl * (oh * logf(t[j] + 1e-12f) + (1.0f - oh) * logf(1.0f - t[j] + 1e-12f));
  }
  atomicAdd(loss, -acc / (float)(M * 20));
}

// ---------------------------------------------------------------------------
// Host orchestration
// ---------------------------------------------------------------------------
static inline size_t P64s(size_t x) { return (x + 63) & ~(size_t)63; }

extern "C" void kernel_launch(void* const* d_in, const int* in_sizes, int n_in,
                              void* d_out, int out_size, void* d_ws, size_t ws_size,
                              hipStream_t stream) {
  (void)in_sizes; (void)n_in; (void)out_size; (void)ws_size;
  const int Nn  = 10000;         // nodes
  const int RP  = 10112;         // 10000 padded to 128 (A-role rows)
  const int KPR = 10048;         // 10000 padded to 64  (K-loop for adj GEMMs)

  // ---- inputs (setup_inputs dict order, params in insertion order) ----
  const float* x     = (const float*)d_in[0];
  const float* adj   = (const float*)d_in[1];
  const float* e1_w  = (const float*)d_in[2];  const float* e1_b = (const float*)d_in[3];
  const float* e2_w  = (const float*)d_in[4];  const float* e2_b = (const float*)d_in[5];
  const float* e3_w  = (const float*)d_in[6];  const float* e3_b = (const float*)d_in[7];
  const float* zl_w  = (const float*)d_in[8];  const float* zl_b = (const float*)d_in[9];
  const float* d1_w  = (const float*)d_in[10]; const float* d1_b = (const float*)d_in[11];
  const float* d2_w  = (const float*)d_in[12]; const float* d2_b = (const float*)d_in[13];
  const float* d3_w  = (const float*)d_in[14]; const float* d3_b = (const float*)d_in[15];
  const float* xb_w  = (const float*)d_in[16]; const float* xb_b = (const float*)d_in[17];
  const float* g0_w  = (const float*)d_in[18];
  const float* g1_w  = (const float*)d_in[19];
  const float* g2_w  = (const float*)d_in[20];
  const float* g3_w  = (const float*)d_in[21];
  const float* gz_w  = (const float*)d_in[22];
  const float* gn_w  = (const float*)d_in[23];
  const float* m1_w  = (const float*)d_in[24]; const float* m1_b  = (const float*)d_in[25];
  const float* m2_w  = (const float*)d_in[26]; const float* m2_b  = (const float*)d_in[27];
  const float* m3_w  = (const float*)d_in[28]; const float* m3_b  = (const float*)d_in[29];
  const float* ml_w  = (const float*)d_in[30]; const float* ml_b  = (const float*)d_in[31];
  const float* mzq_w = (const float*)d_in[32]; const float* mzq_b = (const float*)d_in[33];
  const float* clust = (const float*)d_in[34];

  // ---- output slices (tuple order, flattened) ----
  float* out    = (float*)d_out;
  float* o_xbar = out;                 // 10000*3000
  float* o_q    = out + 30000000;      // 10000*20
  float* o_pred = out + 30200000;      // 10000*20
  float* o_z    = out + 30400000;      // 10000*10
  float* o_nol  = out + 30500000;      // 10000*20
  float* o_loss = out + 30700000;      // 1
  float* o_zF   = out + 30700001;      // 10000*20
  float* o_no   = out + 30900001;      // 10000*500

  // ---- workspace carve-out (256B aligned) ----
  char* ws = (char*)d_ws;
  size_t cur = 0;
  auto alloc = [&](size_t bytes) -> void* {
    void* p = ws + cur;
    cur += (bytes + 255) & ~(size_t)255;
    return p;
  };

  auto cvt = [&](const float* src, int M, int N, unsigned short* dst, int Mp, int Np) {
    size_t tot = (size_t)Mp * Np;
    sdcn_cvt_pad<<<(unsigned)((tot + 255) / 256), 256, 0, stream>>>(src, M, N, dst, Mp, Np);
  };
  // S==1: direct (bias+relu fused).  S>1: split-K atomic accumulate + epilogue pass.
  auto gemm = [&](const unsigned short* A, int lda, const unsigned short* B, int ldb,
                  const float* bias, float* C, int ldc, int N, int Kloop, int relu, int S) {
    if (S <= 1) {
      dim3 grid((unsigned)(P64s((size_t)N) / 64), (unsigned)((Nn + 127) / 128), 1);
      sdcn_gemm_bf16<<<grid, 256, 0, stream>>>(A, lda, B, ldb, bias, C, ldc,
                                               Nn, N, Kloop, relu ? 1 : 0, Kloop);
    } else {
      size_t cn = (size_t)Nn * N;
      sdcn_zerof<<<(unsigned)((cn + 255) / 256), 256, 0, stream>>>(C, cn);
      int chunk = (int)((((size_t)(Kloop + S - 1) / S) + 63) & ~(size_t)63);
      int Sr = (Kloop + chunk - 1) / chunk;
      dim3 grid((unsigned)(P64s((size_t)N) / 64), (unsigned)((Nn + 127) / 128), (unsigned)Sr);
      sdcn_gemm_bf16<<<grid, 256, 0, stream>>>(A, lda, B, ldb, nullptr, C, ldc,
                                               Nn, N, Kloop, 2, chunk);
      if (bias || relu)
        sdcn_bias_act<<<(unsigned)((cn + 255) / 256), 256, 0, stream>>>(C, Nn, N, bias, relu);
    }
  };
  auto cvtW = [&](const float* src, int K, int N) -> unsigned short* {
    int Kp = (int)P64s((size_t)K), Np = (int)P64s((size_t)N);
    unsigned short* dst = (unsigned short*)alloc((size_t)Kp * Np * 2);
    cvt(src, K, N, dst, Kp, Np);
    return dst;
  };
  auto gate = [&](const float* X0, int d0, const float* X1, int d1,
                  const float* X2, int d2, const float* X3, int d3,
                  const float* X4, int d4,
                  const float* W, const float* b, int G, float* P) {
    sdcn_gate_l2<<<(Nn + 7) / 8, 256, 0, stream>>>(X0, d0, X1, d1, X2, d2, X3, d3, X4, d4,
                                                   W, b, G, Nn, P);
  };
  auto mix = [&](const float* P, const float* Z, const float* H, int N,
                 unsigned short* dst, int Mp, int Np) {
    size_t tot = (size_t)Mp * Np;
    sdcn_mix_bf16<<<(unsigned)((tot + 255) / 256), 256, 0, stream>>>(P, Z, H, Nn, N, dst, Mp, Np);
  };

  // bf16 operand pools
  unsigned short* adjB = (unsigned short*)alloc((size_t)RP * KPR * 2);   // A-role, ~L2-resident
  unsigned short* xB   = (unsigned short*)alloc((size_t)RP * 3008 * 2);  // A-role
  unsigned short* sA   = (unsigned short*)alloc((size_t)RP * 2048 * 2);  // A-role scratch
  unsigned short* sB   = (unsigned short*)alloc((size_t)RP * 2048 * 2);  // A-role scratch
  unsigned short* uB   = (unsigned short*)alloc((size_t)KPR * 2048 * 2); // B-role scratch
  unsigned short* netB = (unsigned short*)alloc((size_t)RP * 3072 * 2);  // A-role

  // f32 activations
  float* h1F = (float*)alloc((size_t)Nn * 500  * 4);
  float* h2F = (float*)alloc((size_t)Nn * 500  * 4);
  float* h3F = (float*)alloc((size_t)Nn * 2000 * 4);
  float* z1F = (float*)alloc((size_t)Nn * 500  * 4);
  float* z2F = (float*)alloc((size_t)Nn * 500  * 4);
  float* z3F = (float*)alloc((size_t)Nn * 2000 * 4);
  float* z4F = (float*)alloc((size_t)Nn * 10   * 4);
  float* tF  = (float*)alloc((size_t)Nn * 2000 * 4);
  float* t2F = (float*)alloc((size_t)Nn * 500  * 4);
  float* p1F = (float*)alloc((size_t)Nn * 8 * 4);
  float* p2F = (float*)alloc((size_t)Nn * 8 * 4);
  float* p3F = (float*)alloc((size_t)Nn * 8 * 4);
  float* wF  = (float*)alloc((size_t)Nn * 8 * 4);
  float* pzF = (float*)alloc((size_t)Nn * 8 * 4);

  // ---- convert big inputs + all weights to padded bf16 ----
  cvt(x,   Nn, 3000, xB,   RP, 3008);
  cvt(adj, Nn, Nn,   adjB, RP, KPR);
  unsigned short* e1B = cvtW(e1_w, 3000, 500);
  unsigned short* e2B = cvtW(e2_w, 500,  500);
  unsigned short* e3B = cvtW(e3_w, 500,  2000);
  unsigned short* zlB = cvtW(zl_w, 2000, 10);
  unsigned short* d1B = cvtW(d1_w, 10,   500);
  unsigned short* d2B = cvtW(d2_w, 500,  500);
  unsigned short* d3B = cvtW(d3_w, 500,  2000);
  unsigned short* xbB = cvtW(xb_w, 2000, 3000);
  unsigned short* g0B = cvtW(g0_w, 3000, 500);
  unsigned short* g1B = cvtW(g1_w, 500,  500);
  unsigned short* g2B = cvtW(g2_w, 500,  2000);
  unsigned short* g3B = cvtW(g3_w, 2000, 10);
  unsigned short* gzB = cvtW(gz_w, 3020, 500);
  unsigned short* gnB = cvtW(gn_w, 500,  20);

  // ---- autoencoder ----
  gemm(xB, 3008, e1B, 512, e1_b, h1F, 500, 500, 3008, 1, 1);
  cvt(h1F, Nn, 500, sA, RP, 512);
  gemm(sA, 512, e2B, 512, e2_b, h2F, 500, 500, 512, 1, 1);
  cvt(h2F, Nn, 500, sB, RP, 512);
  gemm(sB, 512, e3B, 2048, e3_b, h3F, 2000, 2000, 512, 1, 1);
  cvt(h3F, Nn, 2000, sA, RP, 2048);
  gemm(sA, 2048, zlB, 64, zl_b, o_z, 10, 10, 2048, 0, 8);        // z -> d_out (split-K)
  cvt(o_z, Nn, 10, sB, RP, 64);
  gemm(sB, 64, d1B, 512, d1_b, tF, 500, 500, 64, 1, 1);
  cvt(tF, Nn, 500, sA, RP, 512);
  gemm(sA, 512, d2B, 512, d2_b, t2F, 500, 500, 512, 1, 1);
  cvt(t2F, Nn, 500, sB, RP, 512);
  gemm(sB, 512, d3B, 2048, d3_b, tF, 2000, 2000, 512, 1, 1);
  cvt(tF, Nn, 2000, sA, RP, 2048);
  gemm(sA, 2048, xbB, 3008, xb_b, o_xbar, 3000, 3000, 2048, 0, 1);  // x_bar -> d_out

  // ---- gated GNN stack (adj is always the A operand) ----
  gemm(xB, 3008, g0B, 512, nullptr, tF, 500, 500, 3008, 0, 1);
  cvt(tF, Nn, 500, uB, KPR, 512);
  gemm(adjB, KPR, uB, 512, nullptr, z1F, 500, 500, KPR, 1, 1);      // z1
  gate(h1F, 500, z1F, 500, nullptr, 0, nullptr, 0, nullptr, 0, m1_w, m1_b, 2, p1F);
  mix(p1F, z1F, h1F, 500, sA, RP, 512);
  gemm(sA, 512, g1B, 512, nullptr, t2F, 500, 500, 512, 0, 1);
  cvt(t2F, Nn, 500, uB, KPR, 512);
  gemm(adjB, KPR, uB, 512, nullptr, z2F, 500, 500, KPR, 1, 1);      // z2
  gate(h2F, 500, z2F, 500, nullptr, 0, nullptr, 0, nullptr, 0, m2_w, m2_b, 2, p2F);
  mix(p2F, z2F, h2F, 500, sA, RP, 512);
  gemm(sA, 512, g2B, 2048, nullptr, tF, 2000, 2000, 512, 0, 1);
  cvt(tF, Nn, 2000, uB, KPR, 2048);
  gemm(adjB, KPR, uB, 2048, nullptr, z3F, 2000, 2000, KPR, 1, 1);   // z3
  gate(h3F, 2000, z3F, 2000, nullptr, 0, nullptr, 0, nullptr, 0, m3_w, m3_b, 2, p3F);
  mix(p3F, z3F, h3F, 2000, sA, RP, 2048);
  gemm(sA, 2048, g3B, 64, nullptr, t2F, 10, 10, 2048, 0, 8);        // split-K
  cvt(t2F, Nn, 10, uB, KPR, 64);
  gemm(adjB, KPR, uB, 64, nullptr, z4F, 10, 10, KPR, 1, 16);        // z4 (split-K + relu)
  gate(z1F, 500, z2F, 500, z3F, 2000, z4F, 10, o_z, 10, ml_w, ml_b, 5, wF);
  {
    size_t tot = (size_t)RP * 3072;
    sdcn_build_net<<<(unsigned)((tot + 255) / 256), 256, 0, stream>>>(
        wF, z1F, z2F, z3F, z4F, o_z, Nn, netB, RP, 3072);
  }
  gemm(netB, 3072, gzB, 512, nullptr, tF, 500, 500, 3072, 0, 1);
  cvt(tF, Nn, 500, uB, KPR, 512);
  gemm(adjB, KPR, uB, 512, nullptr, o_no, 500, 500, KPR, 0, 1);     // net_output -> d_out
  cvt(o_no, Nn, 500, sA, RP, 512);
  gemm(sA, 512, gnB, 64, nullptr, t2F, 20, 20, 512, 0, 4);          // split-K
  cvt(t2F, Nn, 20, uB, KPR, 64);
  gemm(adjB, KPR, uB, 64, nullptr, o_nol, 20, 20, KPR, 0, 16);      // net_output_l (split-K)

  // ---- heads ----
  sdcn_softmax_rows<<<(Nn + 255) / 256, 256, 0, stream>>>(o_nol, o_pred, Nn, 20);
  sdcn_student_t<<<(Nn + 255) / 256, 256, 0, stream>>>(o_z, clust, o_q, Nn);
  gate(o_pred, 20, o_q, 20, nullptr, 0, nullptr, 0, nullptr, 0, mzq_w, mzq_b, 2, pzF);
  sdcn_zero1<<<1, 32, 0, stream>>>(o_loss);
  sdcn_fuse_loss<<<(Nn + 255) / 256, 256, 0, stream>>>(pzF, o_pred, o_q, o_zF, o_loss, Nn);
}
